// Dual_Slef_Attn_03_43576738185511
// MI455X (gfx1250) — compile-verified, matrix operand order checked by reading, not run on hardware
//
#include <hip/hip_runtime.h>
#include <math.h>

// ---------------------------------------------------------------------------
// Types for CDNA5 WMMA (wave32): v16bf A/B fragments, v8f accumulator
// ---------------------------------------------------------------------------
typedef __attribute__((ext_vector_type(16))) __bf16 v16bf;
typedef __attribute__((ext_vector_type(8)))  float  v8f;

union V16BF { v16bf v; __bf16 e[16]; uint4 q[2]; };
union V8F   { v8f   v; float  e[8];  };

constexpr int   Bb   = 128;
constexpr int   Cc   = 512;
constexpr int   Nn0  = 256;
constexpr int   QKd  = 64;
constexpr int   Vd   = 256;
constexpr int   DEG  = 32;
// 1/sqrt(1 + 1e-5): eval-mode BatchNorm with running var = 1
#define RSQ 0.9999950000374997f
#define USE_ASYNC_LDS 1

// LDS tiles are stored K-contiguous per row with stride BK+8 = 40 elements
// (80 B = 5x16 B, so every row base and every 8-element K-run is 16B aligned).
// A lane's 16 fragment elements (ISA 7.12.2) are k = 8*(lane>>4) + {0..7} and
// + {16..23}: exactly two 16-byte LDS loads.
__device__ __forceinline__ v16bf frag_ld(const __bf16* rowp, int half) {
    V16BF f;
    f.q[0] = *(const uint4*)(rowp + half * 8);
    f.q[1] = *(const uint4*)(rowp + half * 8 + 16);
    return f.v;
}

// 32-bit LDS offset of a __shared__ pointer (for async-to-LDS)
__device__ __forceinline__ unsigned lds_addr32(const void* p) {
    return (unsigned)(uintptr_t)(__attribute__((address_space(3))) const void*)p;
}

// pack 4 floats -> 4 bf16 (8 bytes)
__device__ __forceinline__ uint2 pk4bf(float a, float b, float c, float d) {
    __bf16 t[4] = { (__bf16)a, (__bf16)b, (__bf16)c, (__bf16)d };
    return *(const uint2*)t;
}

// ---------------------------------------------------------------------------
// fp32 -> bf16 conversion of x (vectorized: float4 -> 4 x bf16)
// ---------------------------------------------------------------------------
__global__ __launch_bounds__(256) void cvt_f32_bf16(const float* __restrict__ in,
                                                    void* __restrict__ out, long n4) {
    long i = (long)blockIdx.x * 256 + threadIdx.x;
    if (i < n4) {
        float4 v = ((const float4*)in)[i];
        ((uint2*)out)[i] = pk4bf(v.x, v.y, v.z, v.w);
    }
}

// ---------------------------------------------------------------------------
// Generic batched WMMA GEMM:  out[b] = epi( A x B[b] ), tile 64x128, K step 32
//   TA : 0 -> A stored [M,K] (ld=K), 1 -> A stored [K,M] (ld=M)
//   TB : 0 -> B stored [K,N] (ld=N), 1 -> B stored [N,K] (ld=K)
//   ADT/BDT : 0 fp32, 1 bf16  (converted to bf16 while staging through LDS)
//   EPI: 0 none | 1 +beta | 2 bn | 3 bn+relu | 4 bn+resid | 5 bn+resid+relu
//   OUT: 0 f32 [m,n] | 1 bf16 [m,n] | 2 bf16 transposed [n,m] | 3 f32+bf16 dual
// 256 threads = 8 waves; wave w: m-tile (w&3), n-tiles (w>>2)*64 + {0,16,32,48}
// -> 4 v_wmma per wave per K-step sharing one A fragment.
// ---------------------------------------------------------------------------
template<int TA, int TB, int ADT, int BDT, int EPI, int OUT>
__global__ __launch_bounds__(256) void gemm_wmma(
    const void* __restrict__ Ap, const void* __restrict__ Bp,
    void* __restrict__ Op, void* __restrict__ O2p,
    const float* __restrict__ resid,
    const float* __restrict__ gamma, const float* __restrict__ beta,
    int M, int N, int K,
    long sA, long sB, long sO, long sO2, long sR, int ldO)
{
    constexpr int BM = 64, BN = 128, BK = 32, LDT = BK + 8;
    __shared__ __bf16 As[BM][LDT];
    __shared__ __bf16 Bs[BN][LDT];   // transposed: Bs[n][k]

    const int  b    = blockIdx.z;
    const int  m0   = blockIdx.y * BM, n0 = blockIdx.x * BN;
    const int  t    = threadIdx.x, lane = t & 31, w = t >> 5;
    const int  am   = (w & 3) * 16;
    const int  bn0  = (w >> 2) * 64;
    const int  half = lane >> 4, idx16 = lane & 15;
    const long ldA  = TA ? (long)M : (long)K;
    const long ldB  = TB ? (long)K : (long)N;
    const float*  Af = (const float*)Ap;  const __bf16* Ah = (const __bf16*)Ap;
    const float*  Bf = (const float*)Bp;  const __bf16* Bh = (const __bf16*)Bp;
    const long baseA = sA * b, baseB = sB * b;

    v8f acc[4] = {{}, {}, {}, {}};

    for (int k0 = 0; k0 < K; k0 += BK) {
        __syncthreads();
        // ---------------- stage A tile (64 x 32) ----------------
        if constexpr (!TA) {
            // 64 rows x 8 groups of 4 k = 512 slots
            #pragma unroll
            for (int i = 0; i < 2; ++i) {
                int slot = t + i * 256;
                int r = slot >> 3, kc = (slot & 7) * 4;
                long g = baseA + (long)(m0 + r) * ldA + (k0 + kc);
                if constexpr (ADT == 0) {
                    float4 v = *(const float4*)(Af + g);
                    *(uint2*)&As[r][kc] = pk4bf(v.x, v.y, v.z, v.w);
                } else {
#if USE_ASYNC_LDS
                    // bf16 source, layout-preserving: async copy straight to LDS
                    unsigned dst = lds_addr32(&As[r][kc]);
                    asm volatile("global_load_async_to_lds_b64 %0, %1, off"
                                 :: "v"(dst), "v"(Ah + g) : "memory");
#else
                    *(uint2*)&As[r][kc] = *(const uint2*)(Ah + g);
#endif
                }
            }
        } else {
            // 32 rows (k) x 16 groups of 4 m = 512 slots
            #pragma unroll
            for (int i = 0; i < 2; ++i) {
                int slot = t + i * 256;
                int kk = slot >> 4, mc = (slot & 15) * 4;
                long g = baseA + (long)(k0 + kk) * ldA + (m0 + mc);
                if constexpr (ADT == 0) {
                    float4 v = *(const float4*)(Af + g);
                    As[mc + 0][kk] = (__bf16)v.x; As[mc + 1][kk] = (__bf16)v.y;
                    As[mc + 2][kk] = (__bf16)v.z; As[mc + 3][kk] = (__bf16)v.w;
                } else {
                    uint2 raw = *(const uint2*)(Ah + g);
                    const __bf16* rp = (const __bf16*)&raw;
                    As[mc + 0][kk] = rp[0]; As[mc + 1][kk] = rp[1];
                    As[mc + 2][kk] = rp[2]; As[mc + 3][kk] = rp[3];
                }
            }
        }
        // ---------------- stage B tile (32 x 128 -> Bs[n][k]) ----------------
        if constexpr (!TB) {
            // 32 rows (k) x 32 groups of 4 n = 1024 slots
            #pragma unroll
            for (int i = 0; i < 4; ++i) {
                int slot = t + i * 256;
                int kk = slot >> 5, nc = (slot & 31) * 4;
                long g = baseB + (long)(k0 + kk) * ldB + (n0 + nc);
                if constexpr (BDT == 0) {
                    float4 v = *(const float4*)(Bf + g);
                    Bs[nc + 0][kk] = (__bf16)v.x; Bs[nc + 1][kk] = (__bf16)v.y;
                    Bs[nc + 2][kk] = (__bf16)v.z; Bs[nc + 3][kk] = (__bf16)v.w;
                } else {
                    uint2 raw = *(const uint2*)(Bh + g);
                    const __bf16* rp = (const __bf16*)&raw;
                    Bs[nc + 0][kk] = rp[0]; Bs[nc + 1][kk] = rp[1];
                    Bs[nc + 2][kk] = rp[2]; Bs[nc + 3][kk] = rp[3];
                    if (k0 + BK < K)
                        __builtin_prefetch((const void*)(Bh + g + (long)BK * ldB), 0, 3);
                }
            }
        } else {
            // 128 rows (n) x 8 groups of 4 k = 1024 slots
            #pragma unroll
            for (int i = 0; i < 4; ++i) {
                int slot = t + i * 256;
                int nn = slot >> 3, kc = (slot & 7) * 4;
                long g = baseB + (long)(n0 + nn) * ldB + (k0 + kc);
                if constexpr (BDT == 0) {
                    float4 v = *(const float4*)(Bf + g);
                    *(uint2*)&Bs[nn][kc] = pk4bf(v.x, v.y, v.z, v.w);
                    if (k0 + BK < K)
                        __builtin_prefetch((const void*)(Bf + g + BK), 0, 3);
                } else {
                    *(uint2*)&Bs[nn][kc] = *(const uint2*)(Bh + g);
                }
            }
        }
        if constexpr (USE_ASYNC_LDS && ADT == 1 && !TA) {
            asm volatile("s_wait_asynccnt 0x0" ::: "memory");
        }
        __syncthreads();
        // ---------------- fragments + 4 x WMMA ----------------
        v16bf af = frag_ld(&As[am + idx16][0], half);
        #pragma unroll
        for (int j = 0; j < 4; ++j) {
            v16bf bfj = frag_ld(&Bs[bn0 + j * 16 + idx16][0], half);
            acc[j] = __builtin_amdgcn_wmma_f32_16x16x32_bf16(false, af, false, bfj,
                                                             (short)0, acc[j], false, false);
        }
    }

    // ---------------- epilogue ----------------
    #pragma unroll
    for (int j = 0; j < 4; ++j) {
        V8F a8; a8.v = acc[j];
        int nn = n0 + bn0 + j * 16 + idx16;
        #pragma unroll
        for (int r = 0; r < 8; ++r) {
            int mm = m0 + am + half * 8 + r;
            float v = a8.e[r];
            if constexpr (EPI == 1) v += beta[mm];
            if constexpr (EPI >= 2) v = v * (gamma[mm] * RSQ) + beta[mm];
            if constexpr (EPI == 4 || EPI == 5) v += resid[sR * b + (long)mm * ldO + nn];
            if constexpr (EPI == 3 || EPI == 5) v = fmaxf(v, 0.0f);
            if constexpr (OUT == 0) {
                ((float*)Op)[sO * b + (long)mm * ldO + nn] = v;
            } else if constexpr (OUT == 1) {
                ((__bf16*)Op)[sO * b + (long)mm * ldO + nn] = (__bf16)v;
            } else if constexpr (OUT == 2) {
                ((__bf16*)Op)[sO * b + (long)nn * ldO + mm] = (__bf16)v;
            } else {
                ((float*)Op)[sO * b + (long)mm * ldO + nn]  = v;
                ((__bf16*)O2p)[sO2 * b + (long)mm * ldO + nn] = (__bf16)v;
            }
        }
    }
}

// ---------------------------------------------------------------------------
// Row softmax (in place): one wave per row of 256, logits scaled by 1/8
// ---------------------------------------------------------------------------
__global__ __launch_bounds__(256) void softmax_rows(float* __restrict__ S, int rows) {
    int wv   = (int)((blockIdx.x * 256 + threadIdx.x) >> 5);
    int lane = threadIdx.x & 31;
    if (wv >= rows) return;
    float* row = S + (long)wv * 256;
    float vals[8];
    float mx = -1e30f;
    #pragma unroll
    for (int i = 0; i < 8; ++i) { vals[i] = row[lane + 32 * i] * 0.125f; mx = fmaxf(mx, vals[i]); }
    #pragma unroll
    for (int o = 16; o > 0; o >>= 1) mx = fmaxf(mx, __shfl_xor(mx, o, 32));
    float sum = 0.f;
    #pragma unroll
    for (int i = 0; i < 8; ++i) { vals[i] = __expf(vals[i] - mx); sum += vals[i]; }
    #pragma unroll
    for (int o = 16; o > 0; o >>= 1) sum += __shfl_xor(sum, o, 32);
    float inv = 1.0f / sum;
    #pragma unroll
    for (int i = 0; i < 8; ++i) row[lane + 32 * i] = vals[i] * inv;
}

// ---------------------------------------------------------------------------
// SA branch: one wave32 per node; lane = edge (32 contiguous edges per dst).
// Segment softmax over 32 edges, weighted aggregation of node-major bf16 v,
// fused BN + bypass residual; writes channels 256..511 of f (f32 + bf16).
// ---------------------------------------------------------------------------
__global__ __launch_bounds__(256) void sa_edge_kernel(
    const int* __restrict__ esrc,
    const void* __restrict__ qsTp,  // bf16 [B,64,256]
    const void* __restrict__ ksTp,  // bf16 [B,64,256]
    const void* __restrict__ vsNp,  // bf16 [B*N, 256] node-major
    const float* __restrict__ lbsa, // f32  [B,256,256]
    const float* __restrict__ g, const float* __restrict__ be,
    float* __restrict__ f32b, void* __restrict__ fbfp)
{
    __shared__ float sh_a[8][32];
    __shared__ int   sh_s[8][32];
    const __bf16* qsT = (const __bf16*)qsTp;
    const __bf16* ksT = (const __bf16*)ksTp;
    const __bf16* vsN = (const __bf16*)vsNp;
    __bf16* fbf = (__bf16*)fbfp;

    int w = threadIdx.x >> 5, lane = threadIdx.x & 31;
    int node = blockIdx.x * 8 + w;              // 0 .. B*N-1
    int b = node >> 8, n = node & 255;
    long e = (long)node * DEG + lane;
    int src = esrc[e];                          // flat node id (same block)

    const __bf16* qp = qsT + (long)b * QKd * Nn0 + n;
    const __bf16* kp = ksT + (long)(src >> 8) * QKd * Nn0 + (src & 255);
    float dot = 0.f;
    #pragma unroll 8
    for (int qk = 0; qk < QKd; ++qk)
        dot += (float)qp[qk * Nn0] * (float)kp[qk * Nn0];
    float eS = dot * 0.125f;
    float mx = eS;
    #pragma unroll
    for (int o = 16; o > 0; o >>= 1) mx = fmaxf(mx, __shfl_xor(mx, o, 32));
    float ex = __expf(eS - mx);
    float den = ex;
    #pragma unroll
    for (int o = 16; o > 0; o >>= 1) den += __shfl_xor(den, o, 32);
    float a = ex / (den + 1e-16f);
    sh_a[w][lane] = a;
    sh_s[w][lane] = src;
    __syncthreads();

    for (int cb = 0; cb < Vd; cb += 32) {
        int c = cb + lane;
        float acc = 0.f;
        #pragma unroll 8
        for (int j = 0; j < DEG; ++j)
            acc += sh_a[w][j] * (float)vsN[(long)sh_s[w][j] * Vd + c];
        float v = acc * (g[c] * RSQ) + be[c] + lbsa[((long)b * Vd + c) * Nn0 + n];
        long o = ((long)b * Cc + Vd + c) * Nn0 + n;   // channels 256..511
        f32b[o] = v;
        fbf[o]  = (__bf16)v;
    }
}

// ---------------------------------------------------------------------------
// Host orchestration
// ---------------------------------------------------------------------------
extern "C" void kernel_launch(void* const* d_in, const int* in_sizes, int n_in,
                              void* d_out, int out_size, void* d_ws, size_t ws_size,
                              hipStream_t stream) {
    const float* x       = (const float*)d_in[0];
    const int*   eidx    = (const int*)d_in[1];
    const float* Wq_ga   = (const float*)d_in[2];
    const float* Wk_ga   = (const float*)d_in[3];
    const float* Wv_ga   = (const float*)d_in[4];
    const float* bv_ga   = (const float*)d_in[5];
    const float* ga_lb_W = (const float*)d_in[6];
    const float* ga_lb_g = (const float*)d_in[7];
    const float* ga_lb_b = (const float*)d_in[8];
    const float* ga_bn_g = (const float*)d_in[9];
    const float* ga_bn_b = (const float*)d_in[10];
    const float* Wq_sa   = (const float*)d_in[11];
    const float* Wk_sa   = (const float*)d_in[12];
    const float* Wv_sa   = (const float*)d_in[13];
    const float* sa_lb_W = (const float*)d_in[14];
    const float* sa_lb_g = (const float*)d_in[15];
    const float* sa_lb_b = (const float*)d_in[16];
    const float* sa_bn_g = (const float*)d_in[17];
    const float* sa_bn_b = (const float*)d_in[18];
    const float* W1      = (const float*)d_in[19];
    const float* bn1_g   = (const float*)d_in[20];
    const float* bn1_b   = (const float*)d_in[21];
    const float* W2      = (const float*)d_in[22];
    const float* bn2_g   = (const float*)d_in[23];
    const float* bn2_b   = (const float*)d_in[24];
    (void)in_sizes; (void)n_in; (void)out_size; (void)ws_size;

    const long NBX = (long)Bb * Cc * Nn0;      // x elements
    const long QKN = (long)Bb * QKd * Nn0;
    const long VN  = (long)Bb * Vd * Nn0;
    const long FN  = NBX;
    const long HN  = (long)Bb * 4 * Cc * Nn0;

    char* wsp = (char*)d_ws;
    size_t off = 0;
    auto take = [&](size_t bytes) { void* p = wsp + off; off += (bytes + 255) & ~(size_t)255; return p; };
    void*  xb   = take(2 * NBX);
    void*  xqT  = take(2 * QKN);
    void*  xkT  = take(2 * QKN);
    void*  xvT  = take(2 * VN);
    void*  qsT  = take(2 * QKN);
    void*  ksT  = take(2 * QKN);
    void*  vsN  = take(2 * VN);
    float* lbga = (float*)take(4 * VN);
    float* lbsa = (float*)take(4 * VN);
    float* Smat = (float*)take(4 * VN);        // S, then P in place
    float* f32b = (float*)take(4 * FN);
    void*  fbf  = take(2 * FN);
    void*  hbf  = take(2 * HN);

    dim3 blk(256);
    const long sX = (long)Cc * Nn0;            // per-batch x stride

    // 0) x -> bf16
    cvt_f32_bf16<<<dim3((unsigned)(NBX / 4 / 256)), blk, 0, stream>>>(x, xb, NBX / 4);

    dim3 gQK(Nn0 / 128, QKd / 64, Bb);         // (2, 1, 128)
    dim3 gV (Nn0 / 128, Vd  / 64, Bb);         // (2, 4, 128)

    // 1-2) GA q/k projections  (out bf16 [B,64,256])
    gemm_wmma<0,0,0,1,0,1><<<gQK, blk, 0, stream>>>(Wq_ga, xb, xqT, nullptr, nullptr, nullptr, nullptr,
        QKd, Nn0, Cc, 0, sX, (long)QKd * Nn0, 0, 0, Nn0);
    gemm_wmma<0,0,0,1,0,1><<<gQK, blk, 0, stream>>>(Wk_ga, xb, xkT, nullptr, nullptr, nullptr, nullptr,
        QKd, Nn0, Cc, 0, sX, (long)QKd * Nn0, 0, 0, Nn0);
    // 3) GA v projection + bias (out bf16 [B,256,256])
    gemm_wmma<0,0,0,1,1,1><<<gV, blk, 0, stream>>>(Wv_ga, xb, xvT, nullptr, nullptr, nullptr, bv_ga,
        Vd, Nn0, Cc, 0, sX, (long)Vd * Nn0, 0, 0, Nn0);
    // 4-5) SA q/k projections
    gemm_wmma<0,0,0,1,0,1><<<gQK, blk, 0, stream>>>(Wq_sa, xb, qsT, nullptr, nullptr, nullptr, nullptr,
        QKd, Nn0, Cc, 0, sX, (long)QKd * Nn0, 0, 0, Nn0);
    gemm_wmma<0,0,0,1,0,1><<<gQK, blk, 0, stream>>>(Wk_sa, xb, ksT, nullptr, nullptr, nullptr, nullptr,
        QKd, Nn0, Cc, 0, sX, (long)QKd * Nn0, 0, 0, Nn0);
    // 6) SA v projection, node-major transposed store (bf16 [B*N, 256])
    gemm_wmma<0,0,0,1,0,2><<<gV, blk, 0, stream>>>(Wv_sa, xb, vsN, nullptr, nullptr, nullptr, nullptr,
        Vd, Nn0, Cc, 0, sX, (long)Nn0 * Vd, 0, 0, Vd);
    // 7-8) bypass linears with fused BN (f32 [B,256,256])
    gemm_wmma<0,0,0,1,2,0><<<gV, blk, 0, stream>>>(ga_lb_W, xb, lbga, nullptr, nullptr, ga_lb_g, ga_lb_b,
        Vd, Nn0, Cc, 0, sX, (long)Vd * Nn0, 0, 0, Nn0);
    gemm_wmma<0,0,0,1,2,0><<<gV, blk, 0, stream>>>(sa_lb_W, xb, lbsa, nullptr, nullptr, sa_lb_g, sa_lb_b,
        Vd, Nn0, Cc, 0, sX, (long)Vd * Nn0, 0, 0, Nn0);

    // 9) attention scores S[n,m] = sum_q xqT[q,n] * xkT[q,m]  (A transposed)
    dim3 gS(Nn0 / 128, Nn0 / 64, Bb);          // (2, 4, 128)
    gemm_wmma<1,0,1,1,0,0><<<gS, blk, 0, stream>>>(xqT, xkT, Smat, nullptr, nullptr, nullptr, nullptr,
        Nn0, Nn0, QKd, (long)QKd * Nn0, (long)QKd * Nn0, (long)Nn0 * Nn0, 0, 0, Nn0);
    // 10) row softmax (1/8 scale) in place -> P
    softmax_rows<<<dim3(Bb * Nn0 / 8), blk, 0, stream>>>(Smat, Bb * Nn0);
    // 11) ga[v,n] = sum_m xvT[v,m] * P[n,m]  (B transposed, A async-staged) with
    //     fused BN + bypass residual; dual f32/bf16 store into channels 0..255
    gemm_wmma<0,1,1,0,4,3><<<gV, blk, 0, stream>>>(xvT, Smat, f32b, fbf, lbga, ga_bn_g, ga_bn_b,
        Vd, Nn0, Nn0, (long)Vd * Nn0, (long)Nn0 * Nn0, (long)Cc * Nn0, (long)Cc * Nn0,
        (long)Vd * Nn0, Nn0);

    // 12) SA edge attention + aggregation + BN + residual -> channels 256..511
    sa_edge_kernel<<<dim3(Bb * Nn0 / 8), blk, 0, stream>>>(eidx, qsT, ksT, vsN,
        lbsa, sa_bn_g, sa_bn_b, f32b, fbf);

    // 13) MLP: h = relu(bn1(W1 x f))  (out bf16 [B,2048,256])
    dim3 gH(Nn0 / 128, (4 * Cc) / 64, Bb);     // (2, 32, 128)
    gemm_wmma<0,0,0,1,3,1><<<gH, blk, 0, stream>>>(W1, fbf, hbf, nullptr, nullptr, bn1_g, bn1_b,
        4 * Cc, Nn0, Cc, 0, (long)Cc * Nn0, (long)4 * Cc * Nn0, 0, 0, Nn0);
    // 14) out = relu(bn2(W2 x h) + f)  (f32 [B,512,256] -> d_out)
    dim3 gF(Nn0 / 128, Cc / 64, Bb);           // (2, 8, 128)
    gemm_wmma<0,0,0,1,5,0><<<gF, blk, 0, stream>>>(W2, hbf, d_out, nullptr, f32b, bn2_g, bn2_b,
        Cc, Nn0, 4 * Cc, 0, (long)4 * Cc * Nn0, (long)Cc * Nn0, 0, (long)Cc * Nn0, Nn0);
}
